// TorchGraph_56143812493355
// MI455X (gfx1250) — matrix operand order repaired and checked by reading, not compiled, not run on hardware
//
#include <hip/hip_runtime.h>
#include <hip/hip_bf16.h>
#include <math.h>

typedef __attribute__((ext_vector_type(16))) _Float16 v16h;
typedef __attribute__((ext_vector_type(8)))  float    v8f;
typedef unsigned long long u64;

#define NBATCH 8
#define NPB    4096
#define NBTOT  32768     // B*N
#define CIN    48
#define COUT   48
#define KNN    9
#define KXN    64        // padded K for normalized features (48 real + 16 zero)
#define KCAT   96        // concat [nodes | Tx1] K dimension = 2*48 = 3 wmma k-steps

// ---------------- WMMA helpers (CDNA5 gfx1250, wave32) ----------------

__device__ __forceinline__ v8f wmma_acc(v16h a, v16h b, v8f c) {
  // D = A(16x32 f16) * B(32x16 f16) + C(16x16 f32)
  return __builtin_amdgcn_wmma_f32_16x16x32_f16(
      /*neg_a=*/false, a, /*neg_b=*/false, b,
      /*c_mod=*/(short)0, c, /*reuse_a=*/false, /*reuse_b=*/false);
}

// A-matrix 16x32 f16 fragment. p points at (row, k_base) with contiguous K.
// Layout: lanes 0-15 hold K=0..15, lanes 16-31 hold K=16..31 (matches B).
__device__ __forceinline__ v16h load_fragA(const _Float16* p, int half) {
  v16h r;
#pragma unroll
  for (int v = 0; v < 8; ++v) {
    int k = 2 * v + 16 * half;
    r[2 * v]     = p[k];
    r[2 * v + 1] = p[k + 1];
  }
  return r;
}

// B-matrix 32x16 f16 fragment, source contiguous in K (node-major features).
// Layout used here (same as A): lanes 0-15 hold K=0..15, lanes 16-31 hold K=16..31.
__device__ __forceinline__ v16h load_fragB_contig(const _Float16* p, int half) {
  v16h r;
#pragma unroll
  for (int e = 0; e < 16; ++e) r[e] = p[half * 16 + e];
  return r;
}

// B-matrix fragment where source is K-major with row stride 48 (weight matrix).
__device__ __forceinline__ v16h load_fragB_s48(const _Float16* p, int half) {
  v16h r;
#pragma unroll
  for (int e = 0; e < 16; ++e) r[e] = p[(half * 16 + e) * COUT];
  return r;
}

// ---------------- branchless packed-key top-9 ----------------
// key = (ordered_float_bits(score) << 32) | ~index : unsigned-greater == better
// (higher score, tie -> lower index), exactly matching jax.lax.top_k on -dist.
__device__ __forceinline__ u64 make_key(float v, int j) {
  unsigned u = __float_as_uint(v);
  u = (u & 0x80000000u) ? ~u : (u | 0x80000000u);
  return ((u64)u << 32) | (unsigned)(~j);
}
__device__ __forceinline__ int key_index(u64 k) {
  return (int)(~(unsigned)(k & 0xffffffffull));
}
// bk sorted ascending (bk[0] = worst). Fully branchless insert.
__device__ __forceinline__ void insert9(u64 (&bk)[KNN], u64 k) {
  bk[0] = (k > bk[0]) ? k : bk[0];
#pragma unroll
  for (int t = 0; t < KNN - 1; ++t) {
    u64 a = bk[t], b = bk[t + 1];
    bool sw = a > b;
    bk[t]     = sw ? b : a;
    bk[t + 1] = sw ? a : b;
  }
}

// ---------------- kernels ----------------

// Transpose + L2-normalize; emit f32 nodes, split-f16 (hi/lo) raw features
// (first 48 of Acat) and split-f16 normalized features (padded to 64), plus sq.
__global__ void prep_nodes_kernel(const float* __restrict__ x,
                                  float* __restrict__ nodes,
                                  _Float16* __restrict__ acat_h, _Float16* __restrict__ acat_l,
                                  _Float16* __restrict__ xnh, _Float16* __restrict__ xnl,
                                  float* __restrict__ sq) {
  int n = blockIdx.x * blockDim.x + threadIdx.x;
  if (n >= NBTOT) return;
  int b = n >> 12, pix = n & (NPB - 1);
  const float* px = x + (size_t)b * CIN * NPB + pix;
  float v[CIN];
  float ss = 0.f;
#pragma unroll
  for (int c = 0; c < CIN; ++c) { v[c] = px[(size_t)c * NPB]; ss += v[c] * v[c]; }
  float inv = 1.f / fmaxf(sqrtf(ss), 1e-12f);
  float s2 = 0.f;
#pragma unroll
  for (int c = 0; c < CIN; ++c) {
    float raw = v[c];
    nodes[(size_t)n * CIN + c] = raw;
    _Float16 h = (_Float16)raw;
    acat_h[(size_t)n * KCAT + c] = h;
    acat_l[(size_t)n * KCAT + c] = (_Float16)(raw - (float)h);
    float xc = raw * inv;
    s2 += xc * xc;
    _Float16 xh = (_Float16)xc;
    xnh[(size_t)n * KXN + c] = xh;
    xnl[(size_t)n * KXN + c] = (_Float16)(xc - (float)xh);
  }
#pragma unroll
  for (int c = CIN; c < KXN; ++c) {
    xnh[(size_t)n * KXN + c] = (_Float16)0.f;
    xnl[(size_t)n * KXN + c] = (_Float16)0.f;
  }
  sq[n] = s2;
}

// Split-f16 of concatenated weights Wcat[96][48] = [W0; W1].
__global__ void prep_w_kernel(const float* __restrict__ W0, const float* __restrict__ W1,
                              _Float16* __restrict__ wh, _Float16* __restrict__ wl) {
  int t = blockIdx.x * blockDim.x + threadIdx.x;
  if (t >= KCAT * COUT) return;
  int k = t / COUT;
  float w = (k < CIN) ? W0[t] : W1[t - CIN * COUT];
  _Float16 h = (_Float16)w;
  wh[t] = h;
  wl[t] = (_Float16)(w - (float)h);
}

// Fused KNN: per-wave 16-row tile; loop all 256 column tiles of the batch,
// 6 split-f16 WMMAs per 16x16 score tile, LDS transpose, running top-9.
__global__ void __launch_bounds__(256)
knn_topk_kernel(const _Float16* __restrict__ xnh, const _Float16* __restrict__ xnl,
                const float* __restrict__ sq, int* __restrict__ nnidx) {
  __shared__ float s_tile[8][256];
  __shared__ u64   s_mk[8][32 * KNN];
  const int wave = threadIdx.x >> 5;
  const int lane = threadIdx.x & 31;
  const int half = lane >> 4;
  const int lr   = lane & 15;
  const int tileId  = blockIdx.x * 8 + wave;   // 0..2047 (256 tiles per batch)
  const int b       = tileId >> 8;
  const int rowBase = (tileId & 255) << 4;     // row within batch
  const int nodeRow = (b << 12) + rowBase + lr;

  const _Float16* pah = xnh + (size_t)nodeRow * KXN;
  const _Float16* pal = xnl + (size_t)nodeRow * KXN;
  v16h Ah0 = load_fragA(pah, half);
  v16h Ah1 = load_fragA(pah + 32, half);
  v16h Al0 = load_fragA(pal, half);
  v16h Al1 = load_fragA(pal + 32, half);

  u64 bk[KNN];
#pragma unroll
  for (int t = 0; t < KNN; ++t) bk[t] = 0ull;   // strictly below any real key

  const float* sqb = sq + (b << 12);
  for (int ct = 0; ct < 256; ++ct) {
    const int colBase = ct << 4;
    const int nodeCol = (b << 12) + colBase + lr;
    const _Float16* pbh = xnh + (size_t)nodeCol * KXN;
    const _Float16* pbl = xnl + (size_t)nodeCol * KXN;
    v16h Bh0 = load_fragB_contig(pbh, half);
    v16h Bh1 = load_fragB_contig(pbh + 32, half);
    v16h Bl0 = load_fragB_contig(pbl, half);
    v16h Bl1 = load_fragB_contig(pbl + 32, half);
    v8f c = {};
    c = wmma_acc(Ah0, Bh0, c);
    c = wmma_acc(Ah1, Bh1, c);
    c = wmma_acc(Ah0, Bl0, c);
    c = wmma_acc(Ah1, Bl1, c);
    c = wmma_acc(Al0, Bh0, c);
    c = wmma_acc(Al1, Bh1, c);
    __syncthreads();                 // previous tile fully consumed
#pragma unroll
    for (int i = 0; i < 8; ++i)
      s_tile[wave][(i + 8 * half) * 16 + lr] = c[i];
    __syncthreads();
    // lane handles row lr; lanes 0-15 scan cols 0-7, lanes 16-31 cols 8-15
#pragma unroll
    for (int j = 0; j < 8; ++j) {
      int jc = half * 8 + j;
      float dotv = s_tile[wave][lr * 16 + jc];
      int jglob = colBase + jc;                    // within-batch node id
      float score = 2.f * dotv - sqb[jglob];       // = -dist + const(row)
      insert9(bk, make_key(score, jglob));
    }
  }
  // merge the two half-wave partial top-9 lists per row
  __syncthreads();
#pragma unroll
  for (int t = 0; t < KNN; ++t) s_mk[wave][lane * KNN + t] = bk[t];
  __syncthreads();
  if (lane < 16) {
    const int p = lane + 16;
#pragma unroll
    for (int t = 0; t < KNN; ++t) insert9(bk, s_mk[wave][p * KNN + t]);
    const int node = (b << 12) + rowBase + lane;
#pragma unroll
    for (int t = 0; t < KNN; ++t)            // output sorted nearest-first
      nnidx[(size_t)node * KNN + t] = key_index(bk[KNN - 1 - t]);
  }
}

__global__ void deg_zero_kernel(float* __restrict__ deg) {
  int n = blockIdx.x * blockDim.x + threadIdx.x;
  if (n < NBTOT) deg[n] = 0.f;
}

// deg = segment_sum(1, src); the single tail edge (node NB-1, k=8) is dropped.
__global__ void deg_count_kernel(const int* __restrict__ nnidx, float* __restrict__ deg) {
  int n = blockIdx.x * blockDim.x + threadIdx.x;
  if (n >= NBTOT) return;
  int base = n & ~(NPB - 1);
  int kmax = (n == NBTOT - 1) ? (KNN - 1) : KNN;
  for (int k = 0; k < kmax; ++k)
    atomicAdd(&deg[base + nnidx[(size_t)n * KNN + k]], 1.0f);
}

__global__ void dinv_kernel(const float* __restrict__ deg, float* __restrict__ dinv) {
  int n = blockIdx.x * blockDim.x + threadIdx.x;
  if (n >= NBTOT) return;
  float d = deg[n];
  dinv[n] = (d > 0.f) ? rsqrtf(fmaxf(d, 1.f)) : 0.f;
}

// Tx1[n,c] = -dinv[n] * sum_k dinv[src_k] * nodes[src_k, c]; dst unique per node.
// Writes split-f16 into the second half of Acat.
__global__ void tx1_kernel(const int* __restrict__ nnidx, const float* __restrict__ dinv,
                           const float* __restrict__ nodes,
                           _Float16* __restrict__ acat_h, _Float16* __restrict__ acat_l) {
  int t = blockIdx.x * blockDim.x + threadIdx.x;
  if (t >= NBTOT * CIN) return;
  int n = t / CIN, c = t % CIN;
  int base = n & ~(NPB - 1);
  float di = dinv[n];
  int kmax = (n == NBTOT - 1) ? (KNN - 1) : KNN;
  float acc = 0.f;
  for (int k = 0; k < kmax; ++k) {
    int s = base + nnidx[(size_t)n * KNN + k];
    acc += dinv[s] * nodes[(size_t)s * CIN + c];
  }
  float v = -di * acc;
  _Float16 h = (_Float16)v;
  acat_h[(size_t)n * KCAT + CIN + c] = h;
  acat_l[(size_t)n * KCAT + CIN + c] = (_Float16)(v - (float)h);
}

// out = [nodes|Tx1](NBx96) @ [W0;W1](96x48) + bias, split-f16 WMMA, K=3 steps.
__global__ void __launch_bounds__(256)
out_gemm_kernel(const _Float16* __restrict__ acat_h, const _Float16* __restrict__ acat_l,
                const _Float16* __restrict__ wh, const _Float16* __restrict__ wl,
                const float* __restrict__ bias, float* __restrict__ out) {
  const int wave = threadIdx.x >> 5;
  const int lane = threadIdx.x & 31;
  const int half = lane >> 4;
  const int lr   = lane & 15;
  const int tileId  = blockIdx.x * 8 + wave;   // 2048 row tiles of 16
  const int rowBase = tileId << 4;
  const int node    = rowBase + lr;
  const _Float16* pa = acat_h + (size_t)node * KCAT;
  const _Float16* pl = acat_l + (size_t)node * KCAT;
  v16h Ah[3], Al[3];
#pragma unroll
  for (int ks = 0; ks < 3; ++ks) {
    Ah[ks] = load_fragA(pa + ks * 32, half);
    Al[ks] = load_fragA(pl + ks * 32, half);
  }
#pragma unroll
  for (int ctile = 0; ctile < 3; ++ctile) {
    v8f c = {};
#pragma unroll
    for (int ks = 0; ks < 3; ++ks) {
      const _Float16* pbh = wh + (size_t)(ks * 32) * COUT + ctile * 16 + lr;
      const _Float16* pbl = wl + (size_t)(ks * 32) * COUT + ctile * 16 + lr;
      v16h Bh = load_fragB_s48(pbh, half);
      v16h Bl = load_fragB_s48(pbl, half);
      c = wmma_acc(Ah[ks], Bh, c);
      c = wmma_acc(Ah[ks], Bl, c);
      c = wmma_acc(Al[ks], Bh, c);
    }
    float bb = bias[ctile * 16 + lr];
#pragma unroll
    for (int i = 0; i < 8; ++i)
      out[(size_t)(rowBase + i + 8 * half) * COUT + ctile * 16 + lr] = c[i] + bb;
  }
}

// ---------------- host launcher ----------------

extern "C" void kernel_launch(void* const* d_in, const int* in_sizes, int n_in,
                              void* d_out, int out_size, void* d_ws, size_t ws_size,
                              hipStream_t stream) {
  (void)in_sizes; (void)n_in; (void)out_size; (void)ws_size;
  const float* x    = (const float*)d_in[0];
  const float* W0   = (const float*)d_in[1];
  const float* W1   = (const float*)d_in[2];
  const float* bias = (const float*)d_in[3];
  float* out = (float*)d_out;

  char* ws = (char*)d_ws;
  size_t off = 0;
  auto alloc = [&](size_t bytes) -> void* {
    void* p = ws + off;
    off = (off + bytes + 255) & ~(size_t)255;
    return p;
  };
  float*    nodes  = (float*)   alloc((size_t)NBTOT * CIN * 4);
  _Float16* acat_h = (_Float16*)alloc((size_t)NBTOT * KCAT * 2);
  _Float16* acat_l = (_Float16*)alloc((size_t)NBTOT * KCAT * 2);
  _Float16* xnh    = (_Float16*)alloc((size_t)NBTOT * KXN * 2);
  _Float16* xnl    = (_Float16*)alloc((size_t)NBTOT * KXN * 2);
  float*    sq     = (float*)   alloc((size_t)NBTOT * 4);
  int*      nnidx  = (int*)     alloc((size_t)NBTOT * KNN * 4);
  float*    deg    = (float*)   alloc((size_t)NBTOT * 4);
  float*    dinvp  = (float*)   alloc((size_t)NBTOT * 4);
  _Float16* wh     = (_Float16*)alloc((size_t)KCAT * COUT * 2);
  _Float16* wl     = (_Float16*)alloc((size_t)KCAT * COUT * 2);

  prep_nodes_kernel<<<NBTOT / 256, 256, 0, stream>>>(x, nodes, acat_h, acat_l, xnh, xnl, sq);
  prep_w_kernel<<<(KCAT * COUT + 255) / 256, 256, 0, stream>>>(W0, W1, wh, wl);
  knn_topk_kernel<<<256, 256, 0, stream>>>(xnh, xnl, sq, nnidx);
  deg_zero_kernel<<<NBTOT / 256, 256, 0, stream>>>(deg);
  deg_count_kernel<<<NBTOT / 256, 256, 0, stream>>>(nnidx, deg);
  dinv_kernel<<<NBTOT / 256, 256, 0, stream>>>(deg, dinvp);
  tx1_kernel<<<(NBTOT * CIN) / 256, 256, 0, stream>>>(nnidx, dinvp, nodes, acat_h, acat_l);
  out_gemm_kernel<<<256, 256, 0, stream>>>(acat_h, acat_l, wh, wl, bias, out);
}
